// AttentiveResidualDenseBlock_5C_61753039782021
// MI455X (gfx1250) — matrix operand order repaired and checked by reading, not compiled, not run on hardware
//
#include <hip/hip_runtime.h>
#include <hip/hip_bf16.h>

typedef __attribute__((ext_vector_type(16))) _Float16 v16h;
typedef __attribute__((ext_vector_type(8)))  _Float16 v8h;
typedef __attribute__((ext_vector_type(8)))  float    v8f;

#define NB    16      // batch
#define HW    9216    // 96*96
#define PH    98      // padded rows
#define PHW   9604    // 98*98
#define CMAX  320     // total concat channels
#define NEXP  8       // K experts
#define NGUARD 388    // 98*98 - 96*96 guard pixels per sample

// ---------------------------------------------------------------------------
// Zero the 1-pixel halo ring of the padded NHWC activation buffer (all 320 ch)
// so conv tap loads can be unconditional.
// ---------------------------------------------------------------------------
__global__ __launch_bounds__(256) void k_zero_pad(_Float16* __restrict__ act) {
  int idx = blockIdx.x * 256 + threadIdx.x;       // over NB*388*(320/8)
  if (idx >= NB * NGUARD * (CMAX / 8)) return;
  int cg = idx % (CMAX / 8);
  int g  = (idx / (CMAX / 8)) % NGUARD;
  int b  = idx / ((CMAX / 8) * NGUARD);
  int h, w;
  if (g < 98)       { h = 0;  w = g; }
  else if (g < 196) { h = 97; w = g - 98; }
  else { int t = g - 196; h = 1 + (t >> 1); w = (t & 1) ? 97 : 0; }
  v8h z = {};
  *(v8h*)&act[((size_t)(b * PHW + h * PH + w)) * CMAX + cg * 8] = z;
}

// ---------------------------------------------------------------------------
// x [B,64,9216] f32 NCHW -> padded act [B,98,98,320] f16 (interior, ch 0..63)
// ---------------------------------------------------------------------------
__global__ __launch_bounds__(256) void k_nchw_to_nhwc(const float* __restrict__ x,
                                                      _Float16* __restrict__ act) {
  int idx = blockIdx.x * 256 + threadIdx.x;              // over B*64*9216
  if (idx >= NB * 64 * HW) return;
  int p = idx % HW;
  int c = (idx / HW) % 64;
  int b = idx / (HW * 64);
  int h = p / 96, w = p % 96;
  act[((size_t)(b * PHW + (h + 1) * PH + (w + 1))) * CMAX + c] = (_Float16)x[idx];
}

// ---------------------------------------------------------------------------
// pooled[b,c] = mean over the 96x96 interior (one block per (b,c))
// ---------------------------------------------------------------------------
__global__ __launch_bounds__(256) void k_pool(const _Float16* __restrict__ act,
                                              float* __restrict__ pooled, int cin) {
  int c = blockIdx.x % cin;
  int b = blockIdx.x / cin;
  float s = 0.f;
  for (int p = threadIdx.x; p < HW; p += 256) {
    int h = p / 96, w = p % 96;
    s += (float)act[((size_t)(b * PHW + (h + 1) * PH + (w + 1))) * CMAX + c];
  }
  __shared__ float red[256];
  red[threadIdx.x] = s;
  __syncthreads();
  for (int st = 128; st > 0; st >>= 1) {
    if (threadIdx.x < st) red[threadIdx.x] += red[threadIdx.x + st];
    __syncthreads();
  }
  if (threadIdx.x == 0) pooled[b * cin + c] = red[0] * (1.f / (float)HW);
}

// ---------------------------------------------------------------------------
// Attention MLP: relu(pooled@aw1.T+ab1) -> softmax(@aw2.T+ab2) ; bias = attn@bm
// ---------------------------------------------------------------------------
__global__ __launch_bounds__(128) void k_attn(const float* __restrict__ pooled,
                                              const float* __restrict__ aw1,
                                              const float* __restrict__ ab1,
                                              const float* __restrict__ aw2,
                                              const float* __restrict__ ab2,
                                              const float* __restrict__ bm,
                                              float* __restrict__ attn,
                                              float* __restrict__ bias, int cin) {
  int b = blockIdx.x, tid = threadIdx.x;
  int hid = cin / 4;                                    // <= 80
  __shared__ float hsh[80];
  __shared__ float logit[NEXP], eexp[NEXP], prob[NEXP];
  const float* pr = pooled + b * cin;
  if (tid < hid) {
    float s = ab1[tid];
    const float* wr = aw1 + tid * cin;
    for (int c = 0; c < cin; ++c) s += wr[c] * pr[c];
    hsh[tid] = s > 0.f ? s : 0.f;
  }
  __syncthreads();
  if (tid < NEXP) {
    float s = ab2[tid];
    const float* wr = aw2 + tid * hid;
    for (int j = 0; j < hid; ++j) s += wr[j] * hsh[j];
    logit[tid] = s;
  }
  __syncthreads();
  if (tid < NEXP) {
    float mx = logit[0];
    for (int k = 1; k < NEXP; ++k) mx = fmaxf(mx, logit[k]);
    eexp[tid] = __expf(logit[tid] - mx);
  }
  __syncthreads();
  if (tid < NEXP) {
    float den = 0.f;
    for (int k = 0; k < NEXP; ++k) den += eexp[k];
    prob[tid] = eexp[tid] / den;
  }
  __syncthreads();
  if (tid < NEXP) attn[b * NEXP + tid] = prob[tid];
  if (tid < 64) {
    float s = 0.f;
    for (int k = 0; k < NEXP; ++k) s += prob[k] * bm[k * 64 + tid];
    bias[b * 64 + tid] = s;
  }
}

// ---------------------------------------------------------------------------
// Mix experts: wkt[b][rs][o][c] = f16( sum_k attn[b,k] * w[k,o,c,rs] )
// ---------------------------------------------------------------------------
__global__ __launch_bounds__(256) void k_mixw(const float* __restrict__ w,
                                              const float* __restrict__ attn,
                                              _Float16* __restrict__ wkt, int cin) {
  int total = NB * 9 * 64 * cin;
  for (int idx = blockIdx.x * 256 + threadIdx.x; idx < total; idx += gridDim.x * 256) {
    int c  = idx % cin;
    int o  = (idx / cin) % 64;
    int rs = (idx / (cin * 64)) % 9;
    int b  = idx / (cin * 576);
    const float* ab = attn + b * NEXP;
    float s = 0.f;
#pragma unroll
    for (int k = 0; k < NEXP; ++k)
      s += ab[k] * w[((size_t)((k * 64 + o) * cin + c)) * 9 + rs];
    wkt[idx] = (_Float16)s;                             // idx == ((b*9+rs)*64+o)*cin+c
  }
}

// ---------------------------------------------------------------------------
// Implicit-GEMM dynamic conv via v_wmma_f32_16x16x32_f16.
//   Block = 256 thr = 8 waves; one sample, 256 pixels (wave = 32 pixels x 64 M).
//   K-chunk = 32 ch; all 9 taps of a chunk staged in LDS (36 KB) per barrier
//   pair; halo-padded act makes every tap load unconditional.
// ---------------------------------------------------------------------------
template <int CIN, bool FINAL>
__global__ __launch_bounds__(256) void k_dynconv(const _Float16* __restrict__ act,
                                                 const _Float16* __restrict__ wkt,
                                                 const float* __restrict__ bias,
                                                 const float* __restrict__ x_in,
                                                 _Float16* __restrict__ act_out,
                                                 float* __restrict__ out, int chOff) {
  __shared__ __align__(16) _Float16 aLds[9 * 64 * 32];  // 36 KB: tap x out-ch x ch
  const int tid  = threadIdx.x;
  const int lane = tid & 31;
  const int wave = tid >> 5;
  const int half = lane >> 4;                           // 0|1
  const int l16  = lane & 15;
  const int b    = blockIdx.x / 36;
  const int tile = blockIdx.x % 36;
  const int base = tile * 256 + wave * 32;              // 32-run stays in one row
  const int row  = base / 96;
  const int col  = base % 96;

  // Per-lane tap byte-offsets into the padded buffer (all valid thanks to halo).
  const char* actb = (const char*)act;
  unsigned boff[9];
#pragma unroll
  for (int rs = 0; rs < 9; ++rs) {
    const int r = rs / 3, s = rs % 3;                   // padded coords: +1-1 cancels
    boff[rs] = (unsigned)((((b * PHW + (row + r) * PH + (col + l16 + s)) * CMAX)
                           + half * 16) * 2);
  }
  // Weight staging pointer: rs stride (64*CIN elems) folds into imm offsets.
  const _Float16* wsrc = wkt + (size_t)b * 9 * 64 * CIN + (tid >> 2) * CIN + (tid & 3) * 8;
  _Float16* lldst = &aLds[(tid >> 2) * 32 + (tid & 3) * 8];
  const _Float16* llsrcA = &aLds[l16 * 32 + half * 16]; // + mt*512 + rs*2048 imm

  v8f acc[4][2] = {};

  for (int c0 = 0; c0 < CIN; c0 += 32) {
    __syncthreads();
#pragma unroll
    for (int rs = 0; rs < 9; ++rs)                      // stage 9x64x32 weights
      *(v8h*)(lldst + rs * 2048) = *(const v8h*)(wsrc + rs * 64 * CIN);
    __syncthreads();

#pragma unroll
    for (int rs = 0; rs < 9; ++rs) {
      // B fragments: lane element i -> k = half*16+i -> channel c0+half*16+i
      const char* p = actb + boff[rs];
      v8h lo0 = *(const v8h*)p;
      v8h hi0 = *(const v8h*)(p + 16);
      v8h lo1 = *(const v8h*)(p + 16 * CMAX * 2);       // N-tile 1: +16 pixels
      v8h hi1 = *(const v8h*)(p + 16 * CMAX * 2 + 16);
      v16h bf0, bf1;
#pragma unroll
      for (int i = 0; i < 8; ++i) {
        bf0[i] = lo0[i]; bf0[i + 8] = hi0[i];
        bf1[i] = lo1[i]; bf1[i + 8] = hi1[i];
      }
#pragma unroll
      for (int mt = 0; mt < 4; ++mt) {
        // A fragment: row m = l16, k = half*16 + i
        v8h a0 = *(const v8h*)(llsrcA + rs * 2048 + mt * 512);
        v8h a1 = *(const v8h*)(llsrcA + rs * 2048 + mt * 512 + 8);
        v16h af;
#pragma unroll
        for (int i = 0; i < 8; ++i) { af[i] = a0[i]; af[i + 8] = a1[i]; }
        acc[mt][0] = __builtin_amdgcn_wmma_f32_16x16x32_f16(
            false, af, false, bf0, (short)0, acc[mt][0], false, false);
        acc[mt][1] = __builtin_amdgcn_wmma_f32_16x16x32_f16(
            false, af, false, bf1, (short)0, acc[mt][1], false, false);
      }
    }
    wsrc += 32;                                         // next 32-channel chunk
#pragma unroll
    for (int rs = 0; rs < 9; ++rs) boff[rs] += 64;      // +32 f16 channels
  }

  // Epilogue. D layout: lane, vgpr v -> N = l16, M = mt*16 + half*8 + v.
  const float* bv = bias + b * 64;
#pragma unroll
  for (int nt = 0; nt < 2; ++nt) {
    const int w = col + nt * 16 + l16;
    if (!FINAL) {
      _Float16* dst = act_out +
          ((size_t)(b * PHW + (row + 1) * PH + (w + 1))) * CMAX + chOff;
#pragma unroll
      for (int mt = 0; mt < 4; ++mt) {
        v8h o;
#pragma unroll
        for (int v = 0; v < 8; ++v) {
          float val = acc[mt][nt][v] + bv[mt * 16 + half * 8 + v];
          val = val >= 0.f ? val : 0.2f * val;          // leaky relu
          o[v] = (_Float16)val;
        }
        *(v8h*)(dst + mt * 16 + half * 8) = o;          // 8 consecutive channels
      }
    } else {
      const int pix = row * 96 + w;                     // unpadded NCHW index
      float* dst = out + ((size_t)b * 64) * HW + pix;
      const float* xin = x_in + ((size_t)b * 64) * HW + pix;
#pragma unroll
      for (int mt = 0; mt < 4; ++mt)
#pragma unroll
        for (int v = 0; v < 8; ++v) {
          int m = mt * 16 + half * 8 + v;
          dst[(size_t)m * HW] = (acc[mt][nt][v] + bv[m]) * 0.2f + xin[(size_t)m * HW];
        }
    }
  }
}

// ---------------------------------------------------------------------------
extern "C" void kernel_launch(void* const* d_in, const int* in_sizes, int n_in,
                              void* d_out, int out_size, void* d_ws, size_t ws_size,
                              hipStream_t stream) {
  (void)in_sizes; (void)n_in; (void)out_size; (void)ws_size;
  const float* x = (const float*)d_in[0];
  char* ws = (char*)d_ws;

  _Float16* act    = (_Float16*)ws;                 // B*9604*320 f16 = 98,344,960 B
  size_t off = (size_t)NB * PHW * CMAX * 2;
  _Float16* wkt    = (_Float16*)(ws + off); off += (size_t)NB * 9 * 64 * CMAX * 2;
  float*    pooled = (float*)(ws + off);    off += (size_t)NB * CMAX * 4;
  float*    attn   = (float*)(ws + off);    off += (size_t)NB * NEXP * 4 + 224;
  float*    bias   = (float*)(ws + off);
  float*    out    = (float*)d_out;

  {   // zero halo ring, seed channels 0..63 of the interior
    int tg = NB * NGUARD * (CMAX / 8);
    k_zero_pad<<<(tg + 255) / 256, 256, 0, stream>>>(act);
    int total = NB * 64 * HW;
    k_nchw_to_nhwc<<<(total + 255) / 256, 256, 0, stream>>>(x, act);
  }

  const int cins[5] = {64, 128, 192, 256, 320};
  for (int L = 0; L < 5; ++L) {
    int cin = cins[L];
    const float* w   = (const float*)d_in[1 + 6 * L];
    const float* bm  = (const float*)d_in[2 + 6 * L];
    const float* aw1 = (const float*)d_in[3 + 6 * L];
    const float* ab1 = (const float*)d_in[4 + 6 * L];
    const float* aw2 = (const float*)d_in[5 + 6 * L];
    const float* ab2 = (const float*)d_in[6 + 6 * L];

    k_pool<<<NB * cin, 256, 0, stream>>>(act, pooled, cin);
    k_attn<<<NB, 128, 0, stream>>>(pooled, aw1, ab1, aw2, ab2, bm, attn, bias, cin);
    {
      int total  = NB * 9 * 64 * cin;
      int blocks = (total + 255) / 256;
      if (blocks > 4096) blocks = 4096;
      k_mixw<<<blocks, 256, 0, stream>>>(w, attn, wkt, cin);
    }
    const int grid = NB * 36;
    switch (L) {
      case 0: k_dynconv< 64, false><<<grid, 256, 0, stream>>>(act, wkt, bias, nullptr, act, nullptr,  64); break;
      case 1: k_dynconv<128, false><<<grid, 256, 0, stream>>>(act, wkt, bias, nullptr, act, nullptr, 128); break;
      case 2: k_dynconv<192, false><<<grid, 256, 0, stream>>>(act, wkt, bias, nullptr, act, nullptr, 192); break;
      case 3: k_dynconv<256, false><<<grid, 256, 0, stream>>>(act, wkt, bias, nullptr, act, nullptr, 256); break;
      case 4: k_dynconv<320, true ><<<grid, 256, 0, stream>>>(act, wkt, bias, x,       act, out,       0); break;
    }
  }
}
